// MultiHeadAttention_51384988729777
// MI455X (gfx1250) — matrix operand order repaired and checked by reading, not compiled, not run on hardware
//
#include <hip/hip_runtime.h>
#include <stdint.h>

typedef __bf16 bf16_t;
typedef __attribute__((ext_vector_type(16))) __bf16 v16bf;
typedef __attribute__((ext_vector_type(8)))  __bf16 v8bf;
typedef __attribute__((ext_vector_type(8)))  float  v8f;

#define S_LEN   2048
#define DMODEL  1024
#define NHEAD   16
#define DK      64

// ---------------------------------------------------------------------------
// WMMA wrapper: D(f32 16x16) = A(bf16 16x32) * B(bf16 32x16) + C
// ---------------------------------------------------------------------------
__device__ __forceinline__ v8f wmma_bf16(v16bf a, v16bf b, v8f c) {
  return __builtin_amdgcn_wmma_f32_16x16x32_bf16(
      /*neg_a=*/false, a, /*neg_b=*/false, b,
      /*c_mod=*/(short)0, c, /*reuse_a=*/false, /*reuse_b=*/false);
}

__device__ __forceinline__ v16bf pack16(v8bf lo, v8bf hi) {
  v16bf r;
#pragma unroll
  for (int i = 0; i < 8; ++i) { r[i] = lo[i]; r[i + 8] = hi[i]; }
  return r;
}

// A fragment (16x32, MxK) from row-major A, leading dim lda.
__device__ __forceinline__ v16bf load_a_frag(const bf16_t* __restrict__ A,
                                             int lda, int m0, int k0) {
  const int lane = threadIdx.x & 31;
  const int hi = lane >> 4, l = lane & 15;
  const bf16_t* p = A + (size_t)(m0 + l) * lda + k0 + hi * 16;
  v8bf x = *(const v8bf*)p;
  v8bf y = *(const v8bf*)(p + 8);
  return pack16(x, y);
}

// B fragment (32x16, KxN) where B[k][n] = Mm[(n0+n)*ld + k0+k]  (N-major operand)
__device__ __forceinline__ v16bf load_bT_frag(const bf16_t* __restrict__ Mm,
                                              int ld, int n0, int k0) {
  const int lane = threadIdx.x & 31;
  const int hi = lane >> 4, l = lane & 15;
  const bf16_t* p = Mm + (size_t)(n0 + l) * ld + k0 + hi * 16;
  v8bf x = *(const v8bf*)p;
  v8bf y = *(const v8bf*)(p + 8);
  return pack16(x, y);
}

// Async global->LDS 16-byte copy (CDNA5, tracked by ASYNCcnt).
__device__ __forceinline__ void async_b128(bf16_t* lds_ptr, const bf16_t* gptr) {
  unsigned lds_off = (unsigned)(uintptr_t)lds_ptr;  // low 32 bits = LDS offset
  asm volatile("global_load_async_to_lds_b128 %0, %1, off"
               :: "v"(lds_off), "v"(gptr) : "memory");
}

__device__ __forceinline__ void wait_async0() {
  asm volatile("s_wait_asynccnt 0x0" ::: "memory");
}

// Stage a 64-row x 32-K bf16 tile (row stride ld) into LDS [64][32].
// 128 threads x 2 chunks of 16B each = 256 chunks.
__device__ __forceinline__ void stage_tile_async(const bf16_t* __restrict__ g,
                                                 int ld, int r0, int k0,
                                                 bf16_t* sbuf) {
  const int tid = threadIdx.x;
#pragma unroll
  for (int j = 0; j < 2; ++j) {
    const int chunk = tid * 2 + j;
    const int row = chunk >> 2, sub = chunk & 3;
    async_b128(sbuf + row * 32 + sub * 8,
               g + (size_t)(r0 + row) * ld + k0 + sub * 8);
  }
}

// ---------------------------------------------------------------------------
// fp32 -> bf16 conversion (vectorized x4)
// ---------------------------------------------------------------------------
__global__ __launch_bounds__(256) void cvt_f32_bf16(const float* __restrict__ in,
                                                    bf16_t* __restrict__ out, int n) {
  int i = (blockIdx.x * 256 + threadIdx.x) * 4;
  if (i + 3 < n) {
    float4 f = *(const float4*)(in + i);
    out[i + 0] = (bf16_t)f.x;
    out[i + 1] = (bf16_t)f.y;
    out[i + 2] = (bf16_t)f.z;
    out[i + 3] = (bf16_t)f.w;
  }
}

// ---------------------------------------------------------------------------
// fp32 [rows x cols] -> bf16 transposed [cols x rows], 32x32 LDS tiles
// ---------------------------------------------------------------------------
__global__ __launch_bounds__(256) void transpose_cvt_bf16(
    const float* __restrict__ in, bf16_t* __restrict__ out, int rows, int cols) {
  __shared__ float tile[32][33];
  const int bx = blockIdx.x * 32;
  const int by = blockIdx.y * 32;
  const int tx = threadIdx.x & 31;
  const int ty = threadIdx.x >> 5;
#pragma unroll
  for (int i = ty; i < 32; i += 8)
    tile[i][tx] = in[(size_t)(by + i) * cols + bx + tx];
  __syncthreads();
#pragma unroll
  for (int i = ty; i < 32; i += 8)
    out[(size_t)(bx + i) * rows + by + tx] = (bf16_t)tile[tx][i];
}

// ---------------------------------------------------------------------------
// bf16 GEMM, async-LDS double-buffered: C[M,N] = A[M,K] @ Bt^T + bias[N].
// Bt is [N x K] row-major. Block = 128 threads = 4 waves computes a 64x64
// macro-tile; wave w owns rows w*16..w*16+15. A/B k-slabs staged to LDS with
// GLOBAL_LOAD_ASYNC_TO_LDS_B128, double-buffered against WMMA compute.
// Fragment loads are batched (10x ds_load_b128) before the 4 back-to-back
// WMMAs so only one DS wait is needed per k-step.
// ---------------------------------------------------------------------------
__global__ __launch_bounds__(128) void gemm_bf16_kernel(
    const bf16_t* __restrict__ A, const bf16_t* __restrict__ Bt,
    const float* __restrict__ bias, float* __restrict__ Cf,
    bf16_t* __restrict__ Cb, bf16_t* __restrict__ CbT,
    int M, int N, int K) {
  __shared__ bf16_t sA[2][64 * 32];
  __shared__ bf16_t sB[2][64 * 32];

  const int wid  = threadIdx.x >> 5;
  const int lane = threadIdx.x & 31;
  const int hi = lane >> 4, l = lane & 15;
  const int gblocks_n = N >> 6;
  const int gm = blockIdx.x / gblocks_n;
  const int gn = blockIdx.x % gblocks_n;
  const int mB = gm * 64, nB = gn * 64;
  const int m0 = mB + wid * 16;

  v8f acc[4] = {{}, {}, {}, {}};
  const int nk = K / 32;

  // prologue: stage slab 0
  stage_tile_async(A, K, mB, 0, sA[0]);
  stage_tile_async(Bt, K, nB, 0, sB[0]);

  for (int i = 0; i < nk; ++i) {
    const int cur = i & 1, nxt = cur ^ 1;
    wait_async0();       // previous async batch landed (this wave's)
    __syncthreads();     // all waves landed stores; prior readers done
    if (i + 1 < nk) {
      stage_tile_async(A, K, mB, (i + 1) * 32, sA[nxt]);
      stage_tile_async(Bt, K, nB, (i + 1) * 32, sB[nxt]);
    }
    // ---- batch all fragment loads, then 4 back-to-back WMMAs
    const bf16_t* sa = &sA[cur][0];
    const bf16_t* sb = &sB[cur][0];
    const bf16_t* pa = sa + (wid * 16 + l) * 32 + hi * 16;
    v16bf a = pack16(*(const v8bf*)pa, *(const v8bf*)(pa + 8));
    v16bf b0, b1, b2, b3;
    {
      const bf16_t* pb = sb + l * 32 + hi * 16;
      b0 = pack16(*(const v8bf*)(pb + 0 * 16 * 32), *(const v8bf*)(pb + 0 * 16 * 32 + 8));
      b1 = pack16(*(const v8bf*)(pb + 1 * 16 * 32), *(const v8bf*)(pb + 1 * 16 * 32 + 8));
      b2 = pack16(*(const v8bf*)(pb + 2 * 16 * 32), *(const v8bf*)(pb + 2 * 16 * 32 + 8));
      b3 = pack16(*(const v8bf*)(pb + 3 * 16 * 32), *(const v8bf*)(pb + 3 * 16 * 32 + 8));
    }
    acc[0] = wmma_bf16(a, b0, acc[0]);
    acc[1] = wmma_bf16(a, b1, acc[1]);
    acc[2] = wmma_bf16(a, b2, acc[2]);
    acc[3] = wmma_bf16(a, b3, acc[3]);
  }

#pragma unroll
  for (int t = 0; t < 4; ++t) {
    const int col = nB + t * 16 + l;
    const float bv = bias ? bias[col] : 0.0f;
    if (Cf || Cb) {
#pragma unroll
      for (int r = 0; r < 8; ++r) {
        const int row = m0 + r + 8 * hi;
        const float val = acc[t][r] + bv;
        const size_t idx = (size_t)row * N + col;
        if (Cf) Cf[idx] = val;
        if (Cb) Cb[idx] = (bf16_t)val;
      }
    }
    if (CbT) {
      // rows m0+8*hi..+7 = consecutive tokens of one batch -> one 16B store
      const int row0 = m0 + 8 * hi;
      const int bb = row0 / S_LEN;
      const int s0 = row0 - bb * S_LEN;
      v8bf pk;
#pragma unroll
      for (int r = 0; r < 8; ++r) pk[r] = (bf16_t)(acc[t][r] + bv);
      *(v8bf*)(CbT + ((size_t)bb * DMODEL + col) * S_LEN + s0) = pk;
    }
  }
}

// ---------------------------------------------------------------------------
// Flash attention. Q/K: [B, S, D] bf16 (head = 64-col slice).
// Vt: [B, D, S] bf16 (V transposed per batch) -> contiguous P@V B-fragments.
// One wave = 16 query rows for one (b, h). Keys processed 32 at a time.
// V fragments are loaded right after the score WMMAs so their global latency
// overlaps the softmax VALU work.
// ---------------------------------------------------------------------------
__global__ __launch_bounds__(128) void attn_kernel(
    const bf16_t* __restrict__ Qh, const bf16_t* __restrict__ Kh,
    const bf16_t* __restrict__ Vt, bf16_t* __restrict__ Oh) {
  __shared__ bf16_t sP[4][16 * 32];

  const int wid  = threadIdx.x >> 5;
  const int lane = threadIdx.x & 31;
  const int hi = lane >> 4, l = lane & 15;
  const int b = blockIdx.z, h = blockIdx.y;
  const int q0 = (blockIdx.x * 4 + wid) * 16;

  const bf16_t* Qb = Qh + (size_t)b * S_LEN * DMODEL + h * DK;
  const bf16_t* Kb = Kh + (size_t)b * S_LEN * DMODEL + h * DK;
  const bf16_t* Vb = Vt + (size_t)b * DMODEL * S_LEN + (size_t)h * DK * S_LEN;

  const v16bf qa0 = load_a_frag(Qb, DMODEL, q0, 0);
  const v16bf qa1 = load_a_frag(Qb, DMODEL, q0, 32);

  v8f acc0 = {}, acc1 = {}, acc2 = {}, acc3 = {};
  float mrow[8], lrow[8];
#pragma unroll
  for (int r = 0; r < 8; ++r) { mrow[r] = -1e30f; lrow[r] = 0.0f; }

  const float SC = 0.125f;  // 1/sqrt(64)
  bf16_t* sp = &sP[wid][0];

  for (int j0 = 0; j0 < S_LEN; j0 += 32) {
    // ---- batch K^T fragment loads, then 4 score WMMAs
    v16bf kb00 = load_bT_frag(Kb, DMODEL, j0, 0);
    v16bf kb01 = load_bT_frag(Kb, DMODEL, j0, 32);
    v16bf kb10 = load_bT_frag(Kb, DMODEL, j0 + 16, 0);
    v16bf kb11 = load_bT_frag(Kb, DMODEL, j0 + 16, 32);
    v8f s0 = {}, s1 = {};
    s0 = wmma_bf16(qa0, kb00, s0);
    s0 = wmma_bf16(qa1, kb01, s0);
    s1 = wmma_bf16(qa0, kb10, s1);
    s1 = wmma_bf16(qa1, kb11, s1);

    // ---- V fragments: independent of softmax -> issue now, consume later
    v16bf vb0 = load_bT_frag(Vb, S_LEN, 0,  j0);
    v16bf vb1 = load_bT_frag(Vb, S_LEN, 16, j0);
    v16bf vb2 = load_bT_frag(Vb, S_LEN, 32, j0);
    v16bf vb3 = load_bT_frag(Vb, S_LEN, 48, j0);

    // ---- online softmax (rows live in half-wave lane groups)
    float mt[8], p0[8], p1[8], rs[8], sca[8];
#pragma unroll
    for (int r = 0; r < 8; ++r) {
      s0[r] *= SC; s1[r] *= SC;
      mt[r] = fmaxf(s0[r], s1[r]);
    }
#pragma unroll
    for (int off = 1; off < 16; off <<= 1)
#pragma unroll
      for (int r = 0; r < 8; ++r)
        mt[r] = fmaxf(mt[r], __shfl_xor(mt[r], off, 32));
#pragma unroll
    for (int r = 0; r < 8; ++r) {
      const float mn = fmaxf(mrow[r], mt[r]);
      sca[r] = __expf(mrow[r] - mn);
      mrow[r] = mn;
      p0[r] = __expf(s0[r] - mn);
      p1[r] = __expf(s1[r] - mn);
      rs[r] = p0[r] + p1[r];
    }
#pragma unroll
    for (int off = 1; off < 16; off <<= 1)
#pragma unroll
      for (int r = 0; r < 8; ++r)
        rs[r] += __shfl_xor(rs[r], off, 32);
#pragma unroll
    for (int r = 0; r < 8; ++r) {
      lrow[r] = lrow[r] * sca[r] + rs[r];
      acc0[r] *= sca[r]; acc1[r] *= sca[r];
      acc2[r] *= sca[r]; acc3[r] *= sca[r];
    }

    // ---- stage P (16x32 bf16) through LDS: D-layout -> A-fragment
#pragma unroll
    for (int r = 0; r < 8; ++r) {
      sp[(r + 8 * hi) * 32 + l]      = (bf16_t)p0[r];
      sp[(r + 8 * hi) * 32 + 16 + l] = (bf16_t)p1[r];
    }
    asm volatile("s_wait_dscnt 0x0" ::: "memory");  // same-wave DS is in-order
    v8bf plo = *(const v8bf*)(sp + l * 32 + hi * 16);
    v8bf phi = *(const v8bf*)(sp + l * 32 + hi * 16 + 8);
    v16bf pa = pack16(plo, phi);

    // ---- O += P @ V (4 back-to-back WMMAs; V already in registers)
    acc0 = wmma_bf16(pa, vb0, acc0);
    acc1 = wmma_bf16(pa, vb1, acc1);
    acc2 = wmma_bf16(pa, vb2, acc2);
    acc3 = wmma_bf16(pa, vb3, acc3);
  }

  // ---- normalize and store (head-merge implicit via column slice)
#pragma unroll
  for (int r = 0; r < 8; ++r) {
    const float inv = 1.0f / lrow[r];
    const size_t row = (size_t)b * S_LEN + q0 + r + 8 * hi;
    bf16_t* op = Oh + row * DMODEL + h * DK + l;
    op[0]  = (bf16_t)(acc0[r] * inv);
    op[16] = (bf16_t)(acc1[r] * inv);
    op[32] = (bf16_t)(acc2[r] * inv);
    op[48] = (bf16_t)(acc3[r] * inv);
  }
}

// ---------------------------------------------------------------------------
extern "C" void kernel_launch(void* const* d_in, const int* in_sizes, int n_in,
                              void* d_out, int out_size, void* d_ws, size_t ws_size,
                              hipStream_t stream) {
  (void)in_sizes; (void)n_in; (void)out_size; (void)ws_size;

  const float* q   = (const float*)d_in[0];
  const float* k   = (const float*)d_in[1];
  const float* v   = (const float*)d_in[2];
  const float* w_q = (const float*)d_in[3];
  const float* b_q = (const float*)d_in[4];
  const float* w_k = (const float*)d_in[5];
  const float* b_k = (const float*)d_in[6];
  const float* w_v = (const float*)d_in[7];
  const float* b_v = (const float*)d_in[8];
  const float* w_o = (const float*)d_in[9];
  const float* b_o = (const float*)d_in[10];

  const int B = 2, S = S_LEN, D = DMODEL;
  const size_t nAct = (size_t)B * S * D;   // 4,194,304
  const size_t nW   = (size_t)D * D;       // 1,048,576

  char* ws = (char*)d_ws;
  bf16_t* qb  = (bf16_t*)ws; ws += nAct * sizeof(bf16_t);
  bf16_t* kb  = (bf16_t*)ws; ws += nAct * sizeof(bf16_t);
  bf16_t* vb  = (bf16_t*)ws; ws += nAct * sizeof(bf16_t);
  bf16_t* wqt = (bf16_t*)ws; ws += nW * sizeof(bf16_t);   // W^T, bf16
  bf16_t* wkt = (bf16_t*)ws; ws += nW * sizeof(bf16_t);
  bf16_t* wvt = (bf16_t*)ws; ws += nW * sizeof(bf16_t);
  bf16_t* wot = (bf16_t*)ws; ws += nW * sizeof(bf16_t);
  bf16_t* Qh  = (bf16_t*)ws; ws += nAct * sizeof(bf16_t);
  bf16_t* Kh  = (bf16_t*)ws; ws += nAct * sizeof(bf16_t);
  bf16_t* Vt  = (bf16_t*)ws; ws += nAct * sizeof(bf16_t); // [B, D, S]
  bf16_t* Oh  = qb;  // qb dead after Q projection; reuse for attention out

  // ---- fp32 -> bf16 conversions / weight transposes
  cvt_f32_bf16<<<(int)(nAct / 1024), 256, 0, stream>>>(q, qb, (int)nAct);
  cvt_f32_bf16<<<(int)(nAct / 1024), 256, 0, stream>>>(k, kb, (int)nAct);
  cvt_f32_bf16<<<(int)(nAct / 1024), 256, 0, stream>>>(v, vb, (int)nAct);
  dim3 tg(D / 32, D / 32);
  transpose_cvt_bf16<<<tg, 256, 0, stream>>>(w_q, wqt, D, D);
  transpose_cvt_bf16<<<tg, 256, 0, stream>>>(w_k, wkt, D, D);
  transpose_cvt_bf16<<<tg, 256, 0, stream>>>(w_v, wvt, D, D);
  transpose_cvt_bf16<<<tg, 256, 0, stream>>>(w_o, wot, D, D);

  // ---- input projections (64x64 macro-tile blocks)
  const int M = B * S, N = D, K = D;
  const int nblocks = (M / 64) * (N / 64);  // 1024
  gemm_bf16_kernel<<<nblocks, 128, 0, stream>>>(qb, wqt, b_q, nullptr, Qh, nullptr, M, N, K);
  gemm_bf16_kernel<<<nblocks, 128, 0, stream>>>(kb, wkt, b_k, nullptr, Kh, nullptr, M, N, K);
  gemm_bf16_kernel<<<nblocks, 128, 0, stream>>>(vb, wvt, b_v, nullptr, nullptr, Vt, M, N, K);

  // ---- attention (flash, per head column slice; V transposed)
  dim3 ag(S / 64, NHEAD, B);
  attn_kernel<<<ag, 128, 0, stream>>>(Qh, Kh, Vt, Oh);

  // ---- output projection (f32 out)
  gemm_bf16_kernel<<<nblocks, 128, 0, stream>>>(Oh, wot, b_o, (float*)d_out, nullptr, nullptr, M, N, K);
}
